// MultiHeadedAttention_2473901162860
// MI455X (gfx1250) — matrix operand order repaired and checked by reading, not compiled, not run on hardware
//
#include <hip/hip_runtime.h>
#include <hip/hip_bf16.h>

typedef __attribute__((ext_vector_type(16))) _Float16 v16h;
typedef __attribute__((ext_vector_type(8)))  float    v8f;

union V16 {
    v16h  v;
    uint4 q[2];
    _Float16 h[16];
};

#define WMMA_F16(A, B, C) \
    __builtin_amdgcn_wmma_f32_16x16x32_f16(false, (A), false, (B), (short)0, (C), false, false)

// ---------------- f32 -> f16 bulk convert (one-shot; removes cvt from GEMM loops) ----
__global__ __launch_bounds__(256) void cvt_f16_kernel(const float* __restrict__ in,
                                                      _Float16* __restrict__ out, int n) {
    int i = (blockIdx.x * 256 + threadIdx.x) * 8;
    if (i < n) {
        float4 a = *(const float4*)(in + i);
        float4 b = *(const float4*)(in + i + 4);
        V16 o;
        o.h[0] = (_Float16)a.x; o.h[1] = (_Float16)a.y; o.h[2] = (_Float16)a.z; o.h[3] = (_Float16)a.w;
        o.h[4] = (_Float16)b.x; o.h[5] = (_Float16)b.y; o.h[6] = (_Float16)b.z; o.h[7] = (_Float16)b.w;
        *(uint4*)(out + i) = o.q[0];
    }
}

// ---------------- 32x64 register-blocked GEMM core: acc = X(MxK,f16) * W(NxK,f16)^T ----
__device__ __forceinline__ void gemm32x64(const _Float16* __restrict__ X,
                                          const _Float16* __restrict__ W,
                                          int mt, int nt, int lane, int K,
                                          v8f acc[2][4]) {
    const int aoff = (lane >= 16) ? 16 : 0;  // A layout: K range  [aoff..aoff+15]
    const int boff = (lane < 16) ? 0 : 16;   // B layout: K range  [boff..boff+15]
    const _Float16* x0 = X + (size_t)(mt * 32 + (lane & 15)) * K;
    const _Float16* x1 = x0 + (size_t)16 * K;
    const _Float16* w0 = W + (size_t)(nt * 64 + (lane & 15)) * K;
#pragma unroll 2
    for (int kk = 0; kk < K; kk += 32) {
        V16 a0, a1;
        a0.q[0] = *(const uint4*)(x0 + kk + aoff);
        a0.q[1] = *(const uint4*)(x0 + kk + aoff + 8);
        a1.q[0] = *(const uint4*)(x1 + kk + aoff);
        a1.q[1] = *(const uint4*)(x1 + kk + aoff + 8);
#pragma unroll
        for (int j = 0; j < 4; ++j) {
            V16 b;
            const _Float16* wp = w0 + (size_t)(j * 16) * K + kk + boff;
            b.q[0] = *(const uint4*)wp;
            b.q[1] = *(const uint4*)(wp + 8);
            acc[0][j] = WMMA_F16(a0.v, b.v, acc[0][j]);
            acc[1][j] = WMMA_F16(a1.v, b.v, acc[1][j]);
        }
    }
}

// ---------------- QKV projection: Y = X*W^T + b -> f16 [b,h,t,dk] layout, scaled ------
__global__ __launch_bounds__(128) void qkv_proj_kernel(const _Float16* __restrict__ X,
                                                       const _Float16* __restrict__ W,
                                                       const float* __restrict__ bias,
                                                       _Float16* __restrict__ outH,
                                                       float scale) {
    const int lane = threadIdx.x & 31;
    const int wv   = threadIdx.x >> 5;
    const int id   = blockIdx.x * 4 + wv;    // 2048 jobs: 256 mtiles x 8 nstrips
    const int mt   = id >> 3;
    const int nt   = id & 7;

    v8f acc[2][4] = {{{}, {}, {}, {}}, {{}, {}, {}, {}}};
    gemm32x64(X, W, mt, nt, lane, 512, acc);

#pragma unroll
    for (int j = 0; j < 4; ++j) {
        const int col = nt * 64 + j * 16 + (lane & 15);
        const float bc = bias[col];
        const int hh = col >> 6, dd = col & 63;
#pragma unroll
        for (int mi = 0; mi < 2; ++mi)
#pragma unroll
            for (int r = 0; r < 8; ++r) {
                int row = mt * 32 + mi * 16 + r + ((lane >= 16) ? 8 : 0);
                int b = row >> 11, t = row & 2047;
                outH[(((size_t)b * 8 + hh) * 2048 + t) * 64 + dd] =
                    (_Float16)((acc[mi][j][r] + bc) * scale);
            }
    }
}

// ---------------- Output projection: out = X*Wo^T + bo, f32 store --------------------
__global__ __launch_bounds__(128) void out_proj_kernel(const _Float16* __restrict__ X,
                                                       const _Float16* __restrict__ W,
                                                       const float* __restrict__ bias,
                                                       float* __restrict__ out) {
    const int lane = threadIdx.x & 31;
    const int wv   = threadIdx.x >> 5;
    const int id   = blockIdx.x * 4 + wv;
    const int mt   = id >> 3;
    const int nt   = id & 7;

    v8f acc[2][4] = {{{}, {}, {}, {}}, {{}, {}, {}, {}}};
    gemm32x64(X, W, mt, nt, lane, 512, acc);

#pragma unroll
    for (int j = 0; j < 4; ++j) {
        const int col = nt * 64 + j * 16 + (lane & 15);
        const float bc = bias[col];
#pragma unroll
        for (int mi = 0; mi < 2; ++mi)
#pragma unroll
            for (int r = 0; r < 8; ++r) {
                int row = mt * 32 + mi * 16 + r + ((lane >= 16) ? 8 : 0);
                out[(size_t)row * 512 + col] = acc[mi][j][r] + bc;
            }
    }
}

// ---------------- Flash attention: per (b,h,q-tile of 64); 4 waves x 16 q-rows -------
__global__ __launch_bounds__(128) void attn_kernel(const _Float16* __restrict__ Qh,
                                                   const _Float16* __restrict__ Kh,
                                                   const _Float16* __restrict__ Vh,
                                                   const int* __restrict__ mask,
                                                   _Float16* __restrict__ Oh) {
    __shared__ __align__(16) _Float16 Kl[64 * 64];    // [key][d]
    __shared__ __align__(16) _Float16 Vt[64 * 64];    // [d][key] (transposed)
    __shared__ __align__(16) _Float16 Pl[4][16 * 64]; // per-wave P tile [qrow][key]
    __shared__ float mkf[64];

    const int tid  = threadIdx.x;
    const int lane = tid & 31;
    const int wv   = tid >> 5;
    const int bh   = blockIdx.x >> 5;     // b*8 + h
    const int qt   = blockIdx.x & 31;
    const int b    = bh >> 3;
    const int h    = bh & 7;

    const int aoff = (lane >= 16) ? 16 : 0;
    const int boff = (lane < 16) ? 0 : 16;

    // This wave's Q A-operands (16 q-rows x 64 d, Q pre-scaled by 1/sqrt(dk))
    V16 aq[2];
    {
        const _Float16* qrow = Qh + ((size_t)bh * 2048 + qt * 64 + wv * 16 + (lane & 15)) * 64;
#pragma unroll
        for (int ks = 0; ks < 2; ++ks) {
            aq[ks].q[0] = *(const uint4*)(qrow + ks * 32 + aoff);
            aq[ks].q[1] = *(const uint4*)(qrow + ks * 32 + aoff + 8);
        }
    }

    v8f oacc[4] = {{}, {}, {}, {}};
    float m_run[8], l_run[8];
#pragma unroll
    for (int r = 0; r < 8; ++r) { m_run[r] = -3.0e38f; l_run[r] = 0.0f; }

    const _Float16* Kg = Kh + (size_t)bh * 2048 * 64;
    const _Float16* Vg = Vh + (size_t)bh * 2048 * 64;
    const int* mg = mask + (size_t)b * 2048;

#pragma unroll 1
    for (int kb = 0; kb < 2048; kb += 64) {
        __syncthreads();   // previous iteration's LDS consumers done
        // K tile: async global->LDS copies (ASYNCcnt path); V tile: register transpose
#pragma unroll
        for (int j = 0; j < 4; ++j) {
            int c   = j * 128 + tid;      // 512 chunks of 8 f16
            int row = c >> 3;
            int ce  = (c & 7) * 8;
            unsigned ldsK = (unsigned)(unsigned long long)(uintptr_t)&Kl[row * 64 + ce];
            unsigned long long gK = (unsigned long long)(uintptr_t)(Kg + (size_t)(kb + row) * 64 + ce);
            asm volatile("global_load_async_to_lds_b128 %0, %1, off"
                         :: "v"(ldsK), "v"(gK) : "memory");
            uint4 vvq = *(const uint4*)(Vg + (size_t)(kb + row) * 64 + ce);
            const _Float16* vh = (const _Float16*)&vvq;
#pragma unroll
            for (int e = 0; e < 8; ++e) Vt[(ce + e) * 64 + row] = vh[e];
        }
        if (tid < 64) mkf[tid] = (mg[kb + tid] != 0) ? 1.0f : 0.0f;
        asm volatile("s_wait_asynccnt 0" ::: "memory");   // drain this wave's async copies
        __syncthreads();

        // S = Q K^T  (this wave's 16 rows x 64 keys)
        v8f sacc[4];
#pragma unroll
        for (int nt = 0; nt < 4; ++nt) {
            v8f s = {};
#pragma unroll
            for (int ks = 0; ks < 2; ++ks) {
                V16 bk;
                const _Float16* kp = &Kl[(nt * 16 + (lane & 15)) * 64 + ks * 32 + boff];
                bk.q[0] = *(const uint4*)kp;
                bk.q[1] = *(const uint4*)(kp + 8);
                s = WMMA_F16(aq[ks].v, bk.v, s);
            }
            sacc[nt] = s;
        }

        // mask bias + keep flags (per lane: one key column per nt)
        float mkc[4], mbias[4];
#pragma unroll
        for (int nt = 0; nt < 4; ++nt) {
            mkc[nt]   = mkf[nt * 16 + (lane & 15)];
            mbias[nt] = (mkc[nt] - 1.0f) * 1.0e30f;
        }
#pragma unroll
        for (int nt = 0; nt < 4; ++nt)
#pragma unroll
            for (int r = 0; r < 8; ++r) sacc[nt][r] += mbias[nt];

        // online softmax: row max / sum via shuffles within 16-lane column groups
        float mnew[8], lblk[8];
#pragma unroll
        for (int r = 0; r < 8; ++r) {
            float mx = fmaxf(fmaxf(sacc[0][r], sacc[1][r]), fmaxf(sacc[2][r], sacc[3][r]));
#pragma unroll
            for (int d = 1; d < 16; d <<= 1) mx = fmaxf(mx, __shfl_xor(mx, d, 32));
            mnew[r] = fmaxf(m_run[r], mx);
            lblk[r] = 0.0f;
        }
#pragma unroll
        for (int nt = 0; nt < 4; ++nt)
#pragma unroll
            for (int r = 0; r < 8; ++r) {
                float p = __expf(sacc[nt][r] - mnew[r]) * mkc[nt];
                sacc[nt][r] = p;
                lblk[r] += p;
            }
#pragma unroll
        for (int r = 0; r < 8; ++r) {
#pragma unroll
            for (int d = 1; d < 16; d <<= 1) lblk[r] += __shfl_xor(lblk[r], d, 32);
            float alpha = __expf(m_run[r] - mnew[r]);
            l_run[r] = l_run[r] * alpha + lblk[r];
            m_run[r] = mnew[r];
#pragma unroll
            for (int dt = 0; dt < 4; ++dt) oacc[dt][r] *= alpha;
        }

        // P (C/D layout) -> per-wave LDS row-major -> A-operand layout
        _Float16* pw = &Pl[wv][0];
        const int prow_off = (lane >= 16) ? 8 : 0;
#pragma unroll
        for (int nt = 0; nt < 4; ++nt)
#pragma unroll
            for (int r = 0; r < 8; ++r)
                pw[(r + prow_off) * 64 + nt * 16 + (lane & 15)] = (_Float16)sacc[nt][r];
        asm volatile("s_wait_dscnt 0" ::: "memory");   // same-wave LDS RAW

        const _Float16* prow = &Pl[wv][(lane & 15) * 64];
#pragma unroll
        for (int ks = 0; ks < 2; ++ks) {
            V16 ap;
            ap.q[0] = *(const uint4*)(prow + ks * 32 + aoff);
            ap.q[1] = *(const uint4*)(prow + ks * 32 + aoff + 8);
#pragma unroll
            for (int dt = 0; dt < 4; ++dt) {
                V16 bvv;
                const _Float16* vp = &Vt[(dt * 16 + (lane & 15)) * 64 + ks * 32 + boff];
                bvv.q[0] = *(const uint4*)vp;
                bvv.q[1] = *(const uint4*)(vp + 8);
                oacc[dt] = WMMA_F16(ap.v, bvv.v, oacc[dt]);
            }
        }
    }

    // normalize and store f16 to Oh[b, t, h*64+d] (feeds out-projection GEMM)
    const int qrow0 = qt * 64 + wv * 16;
#pragma unroll
    for (int r = 0; r < 8; ++r) {
        int row = qrow0 + r + ((lane >= 16) ? 8 : 0);
        float inv = (l_run[r] > 0.0f) ? (1.0f / l_run[r]) : 0.0f;
#pragma unroll
        for (int dt = 0; dt < 4; ++dt) {
            int f = h * 64 + dt * 16 + (lane & 15);
            Oh[((size_t)b * 2048 + row) * 512 + f] = (_Float16)(oacc[dt][r] * inv);
        }
    }
}

extern "C" void kernel_launch(void* const* d_in, const int* in_sizes, int n_in,
                              void* d_out, int out_size, void* d_ws, size_t ws_size,
                              hipStream_t stream) {
    const float* q    = (const float*)d_in[0];
    const float* k    = (const float*)d_in[1];
    const float* v    = (const float*)d_in[2];
    const int*   msk  = (const int*)d_in[3];
    const float* Wq   = (const float*)d_in[4];
    const float* bq   = (const float*)d_in[5];
    const float* Wk   = (const float*)d_in[6];
    const float* bk   = (const float*)d_in[7];
    const float* Wv   = (const float*)d_in[8];
    const float* bv   = (const float*)d_in[9];
    const float* Wo   = (const float*)d_in[10];
    const float* bo   = (const float*)d_in[11];

    const size_t nX = (size_t)4 * 2048 * 512;   // B*T*F = 4,194,304
    const size_t nW = (size_t)512 * 512;        // 262,144

    _Float16* Xq  = (_Float16*)d_ws;
    _Float16* Xk  = Xq + nX;
    _Float16* Xv  = Xk + nX;
    _Float16* Wqh = Xv + nX;
    _Float16* Wkh = Wqh + nW;
    _Float16* Wvh = Wkh + nW;
    _Float16* Woh = Wvh + nW;
    _Float16* Qh  = Woh + nW;   // [b,h,t,64]
    _Float16* Kh  = Qh + nX;
    _Float16* Vh  = Kh + nX;
    _Float16* Oh  = Vh + nX;    // [b,t,512]

    dim3 blk(128);
    // one-shot f32->f16 conversion
    cvt_f16_kernel<<<(int)(nX / (256 * 8)), 256, 0, stream>>>(q, Xq, (int)nX);
    cvt_f16_kernel<<<(int)(nX / (256 * 8)), 256, 0, stream>>>(k, Xk, (int)nX);
    cvt_f16_kernel<<<(int)(nX / (256 * 8)), 256, 0, stream>>>(v, Xv, (int)nX);
    cvt_f16_kernel<<<(int)(nW / (256 * 8)), 256, 0, stream>>>(Wq, Wqh, (int)nW);
    cvt_f16_kernel<<<(int)(nW / (256 * 8)), 256, 0, stream>>>(Wk, Wkh, (int)nW);
    cvt_f16_kernel<<<(int)(nW / (256 * 8)), 256, 0, stream>>>(Wv, Wvh, (int)nW);
    cvt_f16_kernel<<<(int)(nW / (256 * 8)), 256, 0, stream>>>(Wo, Woh, (int)nW);

    // 8192x512 output, 32x64 tiles per wave, 4 waves/block -> 512 blocks
    qkv_proj_kernel<<<512, blk, 0, stream>>>(Xq, Wqh, bq, Qh, 0.125f); // fold 1/sqrt(64) into Q
    qkv_proj_kernel<<<512, blk, 0, stream>>>(Xk, Wkh, bk, Kh, 1.0f);
    qkv_proj_kernel<<<512, blk, 0, stream>>>(Xv, Wvh, bv, Vh, 1.0f);
    // B*H*(T/64) = 1024 blocks
    attn_kernel<<<1024, blk, 0, stream>>>(Qh, Kh, Vh, msk, Oh);
    out_proj_kernel<<<512, blk, 0, stream>>>(Oh, Woh, bo, (float*)d_out);
}
